// ContextAggregation_13623636263315
// MI455X (gfx1250) — matrix operand order repaired and checked by reading, not compiled, not run on hardware
//
#include <hip/hip_runtime.h>
#include <hip/hip_bf16.h>

typedef __attribute__((ext_vector_type(16))) __bf16 v16bf;
typedef __attribute__((ext_vector_type(8)))  __bf16 v8bf;
typedef __attribute__((ext_vector_type(4)))  __bf16 v4bf;
typedef __attribute__((ext_vector_type(8)))  float  v8f;
typedef __attribute__((ext_vector_type(4)))  float  v4f;

#define NB   4
#define CC   384
#define C3   1152
#define HWN  16384
#define NHD  8
#define HDD  48

// k_gemm tiling
#define BM   64
#define BN   128
#define BK   64
#define LSTR 72            // LDS row stride in halves (144B: 16B-aligned, bank-staggered)
#define BUFSZ (BN * LSTR)  // halves per LDS buffer

#define QK_STRIDE 40       // k_av LDS stride (80B)

__device__ __forceinline__ v16bf cat16(v8bf lo, v8bf hi) {
  return __builtin_shufflevector(lo, hi, 0,1,2,3,4,5,6,7,8,9,10,11,12,13,14,15);
}

__device__ __forceinline__ v8f wmma_bf16(v16bf a, v16bf b, v8f c) {
  return __builtin_amdgcn_wmma_f32_16x16x32_bf16(false, a, false, b, (short)0, c, false, false);
}

// ---------------- f32 -> bf16 conversion (vectorized x4) ----------------
__global__ __launch_bounds__(256) void k_cvt(const float* __restrict__ in,
                                             __bf16* __restrict__ out, long n4) {
  long i = (long)blockIdx.x * blockDim.x + threadIdx.x;
  if (i >= n4) return;
  v4f f = ((const v4f*)in)[i];
  v4bf b;
  #pragma unroll
  for (int j = 0; j < 4; ++j) b[j] = (__bf16)f[j];
  ((v4bf*)out)[i] = b;
}

// ---------------- generic bf16 GEMM: C[m,n] = sum_k A[m,k]*B[k,n] ----------------
// A: [M,K] row-major (batch-invariant). B: [K,N] row-major, batch stride sB.
// OUT_MODE 0: bf16 out, no bias. OUT_MODE 1: f32 out + bias[m].
// Block tile 64(M) x 128(N), K-chunk 64, double-buffered LDS, pipelined loads.
// 8 waves: wave w -> msub = w&3 (rows msub*16..), n-quad = (w>>2)*64, 4 nsubs of 16.
template<int OUT_MODE>
__global__ __launch_bounds__(256) void k_gemm(const __bf16* __restrict__ A,
                                              const __bf16* __restrict__ B,
                                              void* __restrict__ Cv,
                                              const float* __restrict__ bias,
                                              int M, int N, int K,
                                              long sB, long sC) {
  const int n0 = blockIdx.x * BN;
  const int m0 = blockIdx.y * BM;
  const __bf16* Bb = B + (long)blockIdx.z * sB;
  __shared__ __attribute__((aligned(16))) __bf16 bt[2 * BUFSZ];

  const int tid  = threadIdx.x;
  const int lane = tid & 31;
  const int wave = tid >> 5;
  const int msub = wave & 3;
  const int nq   = (wave >> 2) * 64;
  const int lg   = lane >> 4;      // lane group (0/1)
  const int lr   = lane & 15;

  v8f acc[4] = {};
  const __bf16* Arow = A + (long)(m0 + msub * 16 + lr) * K + lg * 8;

  // cooperative stage mapping: thread covers k = kg*4..+4, n = ng*8..+8
  const int kg = tid & 15;         // 16 k-groups of 4
  const int ng = tid >> 4;         // 16 n-groups of 8
  const __bf16* Bg = Bb + (long)(kg * 4) * N + n0 + ng * 8;

  v8bf st[4];
  const int nch = K / BK;
  // prefetch chunk 0
  #pragma unroll
  for (int r = 0; r < 4; ++r) st[r] = *(const v8bf*)(Bg + (long)r * N);

  for (int c = 0; c < nch; ++c) {
    __bf16* buf = bt + (c & 1) * BUFSZ;
    // transposed store: buf[n][k] = B[c*BK + k][n0 + n], b64 granularity
    #pragma unroll
    for (int j = 0; j < 8; ++j) {
      v4bf p = { st[0][j], st[1][j], st[2][j], st[3][j] };
      *(v4bf*)(&buf[(ng * 8 + j) * LSTR + kg * 4]) = p;
    }
    __syncthreads();
    // pipeline: issue next chunk's global loads before computing this chunk
    if (c + 1 < nch) {
      const __bf16* Bn_ = Bg + (long)(c + 1) * BK * N;
      #pragma unroll
      for (int r = 0; r < 4; ++r) st[r] = *(const v8bf*)(Bn_ + (long)r * N);
    }
    #pragma unroll
    for (int ks = 0; ks < BK; ks += 32) {
      const __bf16* ar = Arow + c * BK + ks;
      v16bf af = cat16(*(const v8bf*)ar, *(const v8bf*)(ar + 16));
      #pragma unroll
      for (int j = 0; j < 4; ++j) {
        const __bf16* bp = &buf[(nq + j * 16 + lr) * LSTR + ks + lg * 8];
        v16bf bfr = cat16(*(const v8bf*)bp, *(const v8bf*)(bp + 16));
        acc[j] = wmma_bf16(af, bfr, acc[j]);
      }
    }
    __syncthreads();
  }

  #pragma unroll
  for (int j = 0; j < 4; ++j) {
    const int col = n0 + nq + j * 16 + lr;
    #pragma unroll
    for (int v = 0; v < 8; ++v) {
      const int r = m0 + msub * 16 + lg * 8 + v;
      if (OUT_MODE == 1) {
        float* Co = (float*)Cv + (long)blockIdx.z * sC;
        Co[(long)r * N + col] = acc[j][v] + bias[r];
      } else {
        __bf16* Co = (__bf16*)Cv + (long)blockIdx.z * sC;
        Co[(long)r * N + col] = (__bf16)acc[j][v];
      }
    }
  }
}

// ---------------- row L2 norms of q and k ----------------
// norms[(b*NH+h)*96 + d] : d<48 -> ||q_d||, d>=48 -> ||k_{d-48}||
__global__ __launch_bounds__(256) void k_norms(const __bf16* __restrict__ qkv,
                                               float* __restrict__ norms) {
  const int r  = blockIdx.x;
  const int d  = r % 96;
  const int bh = r / 96;
  const int h  = bh % NHD, b = bh / NHD;
  const int ci = (d < HDD) ? (h * HDD + d) : (CC + h * HDD + (d - HDD));
  const __bf16* row = qkv + (long)b * C3 * HWN + (long)ci * HWN;
  float s = 0.f;
  for (int it = 0; it < HWN; it += 256 * 8) {
    v8bf v = *(const v8bf*)(row + it + threadIdx.x * 8);
    #pragma unroll
    for (int j = 0; j < 8; ++j) { float f = (float)v[j]; s += f * f; }
  }
  __shared__ float red[256];
  red[threadIdx.x] = s; __syncthreads();
  for (int st = 128; st > 0; st >>= 1) {
    if (threadIdx.x < st) red[threadIdx.x] += red[threadIdx.x + st];
    __syncthreads();
  }
  if (threadIdx.x == 0) norms[r] = fmaxf(sqrtf(red[0]), 1e-12f);
}

// ---------------- S = q @ k^T per (b,h): 48x48, K=16384 ----------------
__global__ __launch_bounds__(256) void k_qk(const __bf16* __restrict__ qkv,
                                            float* __restrict__ S) {
  const int bh = blockIdx.x;
  const int h = bh % NHD, b = bh / NHD;
  const __bf16* qb = qkv + (long)b * C3 * HWN + (long)(h * HDD) * HWN;
  const __bf16* kb = qkv + (long)b * C3 * HWN + (long)(CC + h * HDD) * HWN;
  __shared__ float sred[HDD * HDD];

  const int tid = threadIdx.x, lane = tid & 31, wave = tid >> 5;
  const int lg = lane >> 4, lr = lane & 15;
  for (int i = tid; i < HDD * HDD; i += 256) sred[i] = 0.f;
  __syncthreads();

  v8f acc[3][3] = {};
  const long k0 = (long)wave * (HWN / 8);
  for (long kt = k0; kt < k0 + (HWN / 8); kt += 32) {
    v16bf af[3], bfr[3];
    #pragma unroll
    for (int i = 0; i < 3; ++i) {
      const __bf16* ar = qb + (long)(i * 16 + lr) * HWN + kt + lg * 8;
      af[i] = cat16(*(const v8bf*)ar, *(const v8bf*)(ar + 16));
      const __bf16* br = kb + (long)(i * 16 + lr) * HWN + kt + lg * 8;
      bfr[i] = cat16(*(const v8bf*)br, *(const v8bf*)(br + 16));
    }
    #pragma unroll
    for (int i = 0; i < 3; ++i)
      #pragma unroll
      for (int j = 0; j < 3; ++j)
        acc[i][j] = wmma_bf16(af[i], bfr[j], acc[i][j]);
  }
  #pragma unroll
  for (int i = 0; i < 3; ++i)
    #pragma unroll
    for (int j = 0; j < 3; ++j)
      #pragma unroll
      for (int v = 0; v < 8; ++v)
        atomicAdd(&sred[(i * 16 + lg * 8 + v) * HDD + j * 16 + lr], acc[i][j][v]);
  __syncthreads();
  for (int i = tid; i < HDD * HDD; i += 256)
    S[(long)bh * (HDD * HDD) + i] = sred[i];
}

// ---------------- softmax(S * temp / (nq nk)) -> attn bf16 [48][64] (K padded) ----
__global__ __launch_bounds__(256) void k_softmax(const float* __restrict__ S,
                                                 const float* __restrict__ norms,
                                                 const float* __restrict__ temp,
                                                 __bf16* __restrict__ attn) {
  const int bh = blockIdx.x;
  const int h = bh % NHD;
  __shared__ float a[HDD * HDD];
  __shared__ float nq[HDD], nk[HDD];
  const int tid = threadIdx.x;
  if (tid < 96) {
    float v = norms[bh * 96 + tid];
    if (tid < HDD) nq[tid] = v; else nk[tid - HDD] = v;
  }
  __syncthreads();
  const float t = temp[h];
  for (int i = tid; i < HDD * HDD; i += 256) {
    int d = i / HDD, e = i % HDD;
    a[i] = S[(long)bh * (HDD * HDD) + i] * t / (nq[d] * nk[e]);
  }
  __syncthreads();
  if (tid < HDD) {
    float mx = -3.0e38f;
    for (int e = 0; e < HDD; ++e) mx = fmaxf(mx, a[tid * HDD + e]);
    float sm = 0.f;
    for (int e = 0; e < HDD; ++e) { float ex = __expf(a[tid * HDD + e] - mx); a[tid * HDD + e] = ex; sm += ex; }
    float inv = 1.f / sm;
    for (int e = 0; e < HDD; ++e) a[tid * HDD + e] *= inv;
  }
  __syncthreads();
  for (int i = tid; i < HDD * 64; i += 256) {
    int d = i >> 6, e = i & 63;
    attn[(long)bh * (HDD * 64) + i] = (__bf16)((e < HDD) ? a[d * HDD + e] : 0.f);
  }
}

// ---------------- y = attn @ v : per (b,h), M=48, K=64(padded), N-tile 256 ----------
__global__ __launch_bounds__(256) void k_av(const __bf16* __restrict__ qkv,
                                            const __bf16* __restrict__ attn,
                                            __bf16* __restrict__ y) {
  const int b = blockIdx.z, h = blockIdx.y;
  const long t0 = (long)blockIdx.x * 256;
  const __bf16* vb = qkv + (long)b * C3 * HWN + (long)(2 * CC + h * HDD) * HWN;
  const __bf16* ab = attn + (long)(b * NHD + h) * (HDD * 64);
  __bf16* yb = y + (long)b * CC * HWN + (long)(h * HDD) * HWN;
  __shared__ __attribute__((aligned(16))) __bf16 vt[256 * QK_STRIDE];

  const int tid = threadIdx.x, lane = tid & 31, wave = tid >> 5;
  const int lg = lane >> 4, lr = lane & 15;
  v8f acc[3][2] = {};

  for (int s = 0; s < 2; ++s) {
    // stage v^T tile: vt[t_local][e_local] = v[min(s*32+e,47)][t0+t_local]
    const int ee = tid >> 3;
    const int er = (s * 32 + ee < HDD) ? (s * 32 + ee) : (HDD - 1);
    const int tb = (tid & 7) * 32;
    #pragma unroll
    for (int q = 0; q < 4; ++q) {
      v8bf vv = *(const v8bf*)(vb + (long)er * HWN + t0 + tb + q * 8);
      #pragma unroll
      for (int j = 0; j < 8; ++j) vt[(tb + q * 8 + j) * QK_STRIDE + ee] = vv[j];
    }
    __syncthreads();
    #pragma unroll
    for (int i = 0; i < 3; ++i) {
      const __bf16* ar = ab + (long)(i * 16 + lr) * 64 + s * 32 + lg * 8;
      v16bf af = cat16(*(const v8bf*)ar, *(const v8bf*)(ar + 16));
      #pragma unroll
      for (int jn = 0; jn < 2; ++jn) {
        const __bf16* bp = &vt[(wave * 32 + jn * 16 + lr) * QK_STRIDE + lg * 8];
        v16bf bfr = cat16(*(const v8bf*)bp, *(const v8bf*)(bp + 16));
        acc[i][jn] = wmma_bf16(af, bfr, acc[i][jn]);
      }
    }
    __syncthreads();
  }
  #pragma unroll
  for (int i = 0; i < 3; ++i)
    #pragma unroll
    for (int jn = 0; jn < 2; ++jn)
      #pragma unroll
      for (int v = 0; v < 8; ++v)
        yb[(long)(i * 16 + lg * 8 + v) * HWN + t0 + wave * 32 + jn * 16 + lr] =
            (__bf16)acc[i][jn][v];
}

// ============================ host launcher ============================
extern "C" void kernel_launch(void* const* d_in, const int* in_sizes, int n_in,
                              void* d_out, int out_size, void* d_ws, size_t ws_size,
                              hipStream_t stream) {
  const float* x      = (const float*)d_in[0];   // [4,384,128,128]
  const float* qkv_w  = (const float*)d_in[1];   // [1152,384]
  const float* proj_w = (const float*)d_in[2];   // [384,384]
  const float* proj_b = (const float*)d_in[3];   // [384]
  const float* temp   = (const float*)d_in[4];   // [8]

  char* ws = (char*)d_ws;
  size_t off = 0;
  auto take = [&](size_t bytes) -> char* {
    char* p = ws + off;
    off = (off + bytes + 255) & ~(size_t)255;
    return p;
  };
  __bf16* xb    = (__bf16*)take((size_t)NB * CC * HWN * 2);
  __bf16* wq    = (__bf16*)take((size_t)C3 * CC * 2);
  __bf16* wp    = (__bf16*)take((size_t)CC * CC * 2);
  __bf16* qkvb  = (__bf16*)take((size_t)NB * C3 * HWN * 2);
  __bf16* yb    = (__bf16*)take((size_t)NB * CC * HWN * 2);
  float*  Sb    = (float*)take((size_t)NB * NHD * HDD * HDD * 4);
  float*  norms = (float*)take((size_t)NB * NHD * 96 * 4);
  __bf16* attnb = (__bf16*)take((size_t)NB * NHD * HDD * 64 * 2);
  (void)ws_size; (void)n_in; (void)in_sizes; (void)out_size;

  const long n4x = (long)NB * CC * HWN / 4;
  k_cvt<<<dim3((unsigned)((n4x + 255) / 256)), 256, 0, stream>>>(x, xb, n4x);
  k_cvt<<<dim3((C3 * CC / 4 + 255) / 256), 256, 0, stream>>>(qkv_w, wq, C3 * CC / 4);
  k_cvt<<<dim3((CC * CC / 4 + 255) / 256), 256, 0, stream>>>(proj_w, wp, CC * CC / 4);

  // qkv = W_qkv @ x  (per batch), bf16 out
  k_gemm<0><<<dim3(HWN / BN, C3 / BM, NB), 256, 0, stream>>>(
      wq, xb, (void*)qkvb, nullptr, C3, HWN, CC, (long)CC * HWN, (long)C3 * HWN);

  k_norms<<<dim3(NB * NHD * 96), 256, 0, stream>>>(qkvb, norms);
  k_qk<<<dim3(NB * NHD), 256, 0, stream>>>(qkvb, Sb);
  k_softmax<<<dim3(NB * NHD), 256, 0, stream>>>(Sb, norms, temp, attnb);
  k_av<<<dim3(HWN / 256, NHD, NB), 256, 0, stream>>>(qkvb, attnb, yb);

  // out = W_proj @ y + b  (per batch), f32 out
  k_gemm<1><<<dim3(HWN / BN, CC / BM, NB), 256, 0, stream>>>(
      wp, yb, d_out, proj_b, CC, HWN, CC, (long)CC * HWN, (long)CC * HWN);
}
